// ResidualVQ_19722489823683
// MI455X (gfx1250) — compile-verified, matrix operand order checked by reading, not run on hardware
//
#include <hip/hip_runtime.h>

// ---------------------------------------------------------------------------
// ResidualVQ for MI455X (gfx1250): bf16 WMMA everywhere, fused 32-step scan
// kept in LDS per 16-column tile, qout reconstructed as zp - residual_final.
// Round 2: register-block 4 M-tiles per wave in the big projections so each
// (expensive, strided) activation B-fragment feeds 4 WMMAs.
// ---------------------------------------------------------------------------

typedef __attribute__((ext_vector_type(16))) __bf16 bf16x16;
typedef __attribute__((ext_vector_type(8)))  float  f32x8;

#define WMMA_BF16(a, b, c) \
  __builtin_amdgcn_wmma_f32_16x16x32_bf16(false, (a), false, (b), (short)0, (c), false, false)

static constexpr int Bb   = 16;    // batch
static constexpr int Tt   = 2000;  // time steps
static constexpr int DIN  = 1280;
static constexpr int DRVQ = 1024;
static constexpr int NQ   = 32;
static constexpr int KCB  = 1024;  // codebook entries
static constexpr int DCB  = 8;     // codebook dim
static constexpr int NT   = 125;   // Tt / 16

// Load 8 contiguous bf16 (16B, aligned) into elements [BASE, BASE+8) of a frag.
template <int BASE>
__device__ __forceinline__ void load8bf(const __bf16* p, bf16x16& f) {
  union U { uint4 u; __bf16 b[8]; } cv;
  cv.u = *(const uint4*)p;
#pragma unroll
  for (int j = 0; j < 8; ++j) f[BASE + j] = cv.b[j];
}

__device__ __forceinline__ unsigned ordkey(float f) {
  unsigned u = __float_as_uint(f);
  return (u & 0x80000000u) ? ~u : (u | 0x80000000u);
}

// ---------------------------------------------------------------------------
// Weight-norm prep: scale = g[r]/||v_r||, emit bf16 rows.
// ---------------------------------------------------------------------------
__global__ __launch_bounds__(256)
void wn_rows_bf16(const float* __restrict__ v, const float* __restrict__ g,
                  __bf16* __restrict__ dst, int C) {
  __shared__ float sred[256];
  const int r = blockIdx.x, tid = threadIdx.x;
  const float* row = v + (size_t)r * C;
  float s = 0.f;
  for (int c = tid; c < C; c += 256) { float x = row[c]; s += x * x; }
  sred[tid] = s;
  __syncthreads();
  for (int st = 128; st > 0; st >>= 1) {
    if (tid < st) sred[tid] += sred[tid + st];
    __syncthreads();
  }
  const float sc = g[r] * rsqrtf(sred[0]);
  for (int c = tid; c < C; c += 256) dst[(size_t)r * C + c] = (__bf16)(row[c] * sc);
}

__global__ __launch_bounds__(256)
void wn_rows8_bf16(const float* __restrict__ v, const float* __restrict__ g,
                   __bf16* __restrict__ dst, int R) {
  const int r = blockIdx.x * blockDim.x + threadIdx.x;
  if (r >= R) return;
  float s = 0.f;
#pragma unroll
  for (int j = 0; j < 8; ++j) { float x = v[(size_t)r * 8 + j]; s += x * x; }
  const float sc = g[r] * rsqrtf(s);
#pragma unroll
  for (int j = 0; j < 8; ++j) dst[(size_t)r * 8 + j] = (__bf16)(v[(size_t)r * 8 + j] * sc);
}

// Codebook prep: transposed bf16 copy [q][d][k] + per-entry squared norms.
__global__ __launch_bounds__(256)
void cb_prep(const float* __restrict__ cb, __bf16* __restrict__ cbT,
             float* __restrict__ cbn2) {
  const int e = blockIdx.x * blockDim.x + threadIdx.x;  // over NQ*KCB
  if (e >= NQ * KCB) return;
  const int q = e >> 10, k = e & 1023;
  float s = 0.f;
#pragma unroll
  for (int d = 0; d < 8; ++d) {
    float x = cb[(size_t)e * DCB + d];
    s += x * x;
    cbT[(size_t)q * DCB * KCB + (size_t)d * KCB + k] = (__bf16)x;
  }
  cbn2[e] = s;
}

// ---------------------------------------------------------------------------
// Big projections: Y[b,m,t] = sum_k W[m,k] * X[b,k,t] + bias[m]
// 8 waves/block; each wave owns FOUR 16x16 C tiles stacked in M, sharing one
// activation B fragment per K chunk (4 WMMAs per strided B load burst).
// Grid: (M/16/32, T/16, B) = (2, 125, 16) for M=1024.
// ---------------------------------------------------------------------------
__global__ __launch_bounds__(256)
void gemm_wn_bf16(const __bf16* __restrict__ W, const float* __restrict__ X,
                  const float* __restrict__ bias, float* __restrict__ Y, int K) {
  const int lane = threadIdx.x & 31, w = threadIdx.x >> 5;
  const int n = lane & 15, h = lane >> 4;
  const int mt0 = (blockIdx.x * 8 + w) * 4;   // first of 4 M-tiles
  const int col = blockIdx.y * 16 + n;
  const int b   = blockIdx.z;
  const float* Xb = X + (size_t)b * K * Tt;

  f32x8 acc[4] = {};
  for (int k0 = 0; k0 < K; k0 += 32) {
    // One activation fragment, shared by all 4 M-tiles of this wave.
    bf16x16 bm;
    const float* xp = Xb + (size_t)(k0 + 8 * h) * Tt + col;
#pragma unroll
    for (int j = 0; j < 8; ++j) bm[j] = (__bf16)xp[(size_t)j * Tt];
#pragma unroll
    for (int j = 0; j < 8; ++j) bm[8 + j] = (__bf16)xp[(size_t)(16 + j) * Tt];

    const __bf16* ap = W + (size_t)(mt0 * 16 + n) * K + k0 + 8 * h;
    if (k0 + 32 < K) __builtin_prefetch(ap + 32, 0, 1);  // global_prefetch_b8

    bf16x16 a[4];
#pragma unroll
    for (int i = 0; i < 4; ++i) {
      const __bf16* api = ap + (size_t)i * 16 * K;
      load8bf<0>(api, a[i]);
      load8bf<8>(api + 16, a[i]);
    }
#pragma unroll
    for (int i = 0; i < 4; ++i) acc[i] = WMMA_BF16(a[i], bm, acc[i]);
  }

  float* Yb = Y + (size_t)b * DRVQ * Tt;
#pragma unroll
  for (int i = 0; i < 4; ++i) {
#pragma unroll
    for (int r = 0; r < 8; ++r) {
      const int m = (mt0 + i) * 16 + r + 8 * h;
      Yb[(size_t)m * Tt + col] = acc[i][r] + bias[m];
    }
  }
}

// ---------------------------------------------------------------------------
// Fused 32-quantizer scan. One block owns 16 columns of one batch; residual
// lives in LDS fp32. Per quantizer: z_e via K-split WMMA (deterministic LDS
// partial combine), distance WMMA + ds_min_u64 argmin, commit tree-reduce,
// rank-8 residual update via WMMA. Finally act <- zp - residual (= qout).
// ---------------------------------------------------------------------------
__global__ __launch_bounds__(256)
void rvq_fused(const int* __restrict__ input_length,
               const __bf16* __restrict__ WqIn,   // [NQ][8][DRVQ]
               const float* __restrict__ qInB,    // [NQ][8]
               const __bf16* __restrict__ WqOut,  // [NQ][DRVQ][8]
               const float* __restrict__ qOutB,   // [NQ][DRVQ]
               const float* __restrict__ cb,      // [NQ][KCB][8]
               const __bf16* __restrict__ cbT,    // [NQ][8][KCB]
               const float* __restrict__ cbn2,    // [NQ][KCB]
               float* __restrict__ act,           // in: zp, out: qout (in place)
               float* __restrict__ idxOut,        // [NQ][B][T] (as float)
               float* __restrict__ commitPart)    // [NQ][B][NT]
{
  __shared__ float res[DRVQ * 16];                 // 64 KB residual tile [i][t]
  __shared__ float zebuf[16 * 16];                 // z_e as [d][t] (rows>=8 are 0)
  __shared__ float pbuf[8 * 16 * 16];              // per-wave z_e partials
  __shared__ unsigned long long minbuf[16];        // packed (ordfloat, idx) per t
  __shared__ float redbuf[256];

  const int tile = blockIdx.x, b = blockIdx.y;
  const int t0 = tile * 16;
  const int tid = threadIdx.x, lane = tid & 31, w = tid >> 5;
  const int n = lane & 15, h = lane >> 4;
  float* actb = act + (size_t)b * DRVQ * Tt;

  for (int e = tid; e < DRVQ * 16; e += 256)
    res[e] = actb[(size_t)(e >> 4) * Tt + t0 + (e & 15)];
  const float maskcol = (t0 + n < input_length[b]) ? 1.f : 0.f;
  __syncthreads();

  for (int q = 0; q < NQ; ++q) {
    // ---- z_e = WqIn[q] @ (res * mask) : K split 8 ways across waves ----
    f32x8 acc = {};
    const __bf16* Aq = WqIn + (size_t)q * 8 * DRVQ;
#pragma unroll
    for (int kc = 0; kc < 4; ++kc) {
      const int k0 = w * 128 + kc * 32;
      bf16x16 a;
      if (n < 8) {
        const __bf16* ap = Aq + (size_t)n * DRVQ + k0 + 8 * h;
        load8bf<0>(ap, a);
        load8bf<8>(ap + 16, a);
      } else {
#pragma unroll
        for (int j = 0; j < 16; ++j) a[j] = (__bf16)0.f;
      }
      bf16x16 bm;
      const int kb = k0 + 8 * h;
#pragma unroll
      for (int j = 0; j < 8; ++j) bm[j] = (__bf16)(res[(kb + j) * 16 + n] * maskcol);
#pragma unroll
      for (int j = 0; j < 8; ++j) bm[8 + j] = (__bf16)(res[(16 + kb + j) * 16 + n] * maskcol);
      acc = WMMA_BF16(a, bm, acc);
    }
#pragma unroll
    for (int r = 0; r < 8; ++r) pbuf[w * 256 + (r + 8 * h) * 16 + n] = acc[r];
    __syncthreads();
    {
      const int d = tid >> 4, t = tid & 15;
      float v = (d < 8) ? qInB[q * 8 + d] : 0.f;
#pragma unroll
      for (int ww = 0; ww < 8; ++ww) v += pbuf[ww * 256 + d * 16 + t];
      zebuf[d * 16 + t] = v;  // fixed-order sum -> deterministic
    }
    if (tid < 16) minbuf[tid] = ~0ULL;
    __syncthreads();

    // ---- distances: d2 = cbn2[k] - 2 * (z_e . cb_k); argmin via ds_min_u64 ----
    bf16x16 a2;
    {
      const int kb = 8 * h;
#pragma unroll
      for (int j = 0; j < 8; ++j) a2[j] = (__bf16)zebuf[(kb + j) * 16 + n];
#pragma unroll
      for (int j = 0; j < 8; ++j) a2[8 + j] = (__bf16)0.f;
    }
#pragma unroll
    for (int c = 0; c < 8; ++c) {
      const int kcol = (w * 8 + c) * 16 + n;
      bf16x16 b2;
#pragma unroll
      for (int j = 0; j < 8; ++j) {
        __bf16 v = (__bf16)0.f;
        if (h == 0) v = cbT[(size_t)q * DCB * KCB + (size_t)j * KCB + kcol];
        b2[j] = v;
      }
#pragma unroll
      for (int j = 0; j < 8; ++j) b2[8 + j] = (__bf16)0.f;
      f32x8 c2 = {};
      c2 = WMMA_BF16(a2, b2, c2);
      const float c2n = cbn2[q * KCB + kcol];
#pragma unroll
      for (int r = 0; r < 8; ++r) {
        const float d2 = c2n - 2.f * c2[r];
        const unsigned long long key =
            ((unsigned long long)ordkey(d2) << 32) | (unsigned)kcol;
        atomicMin(&minbuf[r + 8 * h], key);  // -> ds_min_u64; order-invariant
      }
    }
    __syncthreads();

    // ---- indices + commit loss ----
    if (tid < 16)
      idxOut[((size_t)q * Bb + b) * Tt + t0 + tid] =
          (float)(unsigned)(minbuf[tid] & 0xFFFFFFFFull);
    float cv = 0.f;
    if (tid < 128) {
      const int t = tid & 15, d = tid >> 4;
      const unsigned kidx = (unsigned)(minbuf[t] & 0xFFFFFFFFull);
      const float diff = zebuf[d * 16 + t] - cb[((size_t)q * KCB + kidx) * DCB + d];
      cv = diff * diff;
    }
    redbuf[tid] = cv;
    __syncthreads();
    for (int s = 128; s > 0; s >>= 1) {
      if (tid < s) redbuf[tid] += redbuf[tid + s];
      __syncthreads();
    }
    if (tid == 0) commitPart[((size_t)q * Bb + b) * NT + tile] = redbuf[0];

    // ---- residual -= WqOut[q] @ z_q + ob ----
    bf16x16 b3;
    {
      const unsigned kidxn = (unsigned)(minbuf[n] & 0xFFFFFFFFull);
#pragma unroll
      for (int j = 0; j < 8; ++j) {
        __bf16 v = (__bf16)0.f;
        if (h == 0) v = (__bf16)cb[((size_t)q * KCB + kidxn) * DCB + j];
        b3[j] = v;
      }
#pragma unroll
      for (int j = 0; j < 8; ++j) b3[8 + j] = (__bf16)0.f;
    }
#pragma unroll
    for (int c = 0; c < 8; ++c) {
      const int row0 = (w * 8 + c) * 16;
      bf16x16 a3;
#pragma unroll
      for (int j = 0; j < 16; ++j) a3[j] = (__bf16)0.f;
      if (h == 0) load8bf<0>(WqOut + ((size_t)q * DRVQ + row0 + n) * DCB, a3);
      f32x8 c3 = {};
      c3 = WMMA_BF16(a3, b3, c3);
#pragma unroll
      for (int r = 0; r < 8; ++r) {
        const int o = row0 + r + 8 * h;
        res[o * 16 + n] -= (c3[r] + qOutB[q * DRVQ + o]);
      }
    }
    __syncthreads();
  }

  // qout = zp - residual_final, written in place over the activation buffer.
  for (int e = tid; e < DRVQ * 16; e += 256) {
    const size_t g = (size_t)(e >> 4) * Tt + t0 + (e & 15);
    actb[g] = actb[g] - res[e];
  }
}

__global__ __launch_bounds__(256)
void commit_reduce(const float* __restrict__ part, float* __restrict__ out) {
  const int e = blockIdx.x * blockDim.x + threadIdx.x;  // over NQ*Bb
  if (e >= NQ * Bb) return;
  float s = 0.f;
  for (int i = 0; i < NT; ++i) s += part[(size_t)e * NT + i];  // fixed order
  out[e] = s * (1.f / ((float)Tt * (float)DCB));
}

// ---------------------------------------------------------------------------
extern "C" void kernel_launch(void* const* d_in, const int* in_sizes, int n_in,
                              void* d_out, int out_size, void* d_ws, size_t ws_size,
                              hipStream_t stream) {
  (void)in_sizes; (void)n_in; (void)out_size; (void)ws_size;
  const float* z     = (const float*)d_in[0];
  const int*   ilen  = (const int*)  d_in[1];
  const float* inV   = (const float*)d_in[2];
  const float* inG   = (const float*)d_in[3];
  const float* inB   = (const float*)d_in[4];
  const float* outV  = (const float*)d_in[5];
  const float* outG  = (const float*)d_in[6];
  const float* outB  = (const float*)d_in[7];
  const float* qInV  = (const float*)d_in[8];
  const float* qInG  = (const float*)d_in[9];
  const float* qInB  = (const float*)d_in[10];
  const float* qOutV = (const float*)d_in[11];
  const float* qOutG = (const float*)d_in[12];
  const float* qOutB = (const float*)d_in[13];
  const float* cb    = (const float*)d_in[14];
  // d_in[15] = n_quantizers (compile-time NQ=32 here)

  char* p = (char*)d_ws;
  auto alloc = [&](size_t bytes) -> char* {
    char* r = p;
    p += (bytes + 255) & ~(size_t)255;
    return r;
  };
  __bf16* WinBf   = (__bf16*)alloc((size_t)DRVQ * DIN * 2);
  __bf16* WoutBf  = (__bf16*)alloc((size_t)DRVQ * DRVQ * 2);
  __bf16* WqInBf  = (__bf16*)alloc((size_t)NQ * 8 * DRVQ * 2);
  __bf16* WqOutBf = (__bf16*)alloc((size_t)NQ * DRVQ * 8 * 2);
  __bf16* cbT     = (__bf16*)alloc((size_t)NQ * DCB * KCB * 2);
  float*  cbn2    = (float*) alloc((size_t)NQ * KCB * 4);
  float*  cpart   = (float*) alloc((size_t)NQ * Bb * NT * 4);
  float*  act     = (float*) alloc((size_t)Bb * DRVQ * Tt * 4);

  float* out  = (float*)d_out;
  float* idxo = out + (size_t)Bb * DRVQ * Tt;          // all_idx as float
  float* como = idxo + (size_t)NQ * Bb * Tt;           // all_commit

  // Weight-norm prep (bf16 for WMMA operands)
  wn_rows_bf16 <<<DRVQ,        256, 0, stream>>>(inV,  inG,  WinBf,  DIN);
  wn_rows_bf16 <<<DRVQ,        256, 0, stream>>>(outV, outG, WoutBf, DRVQ);
  wn_rows_bf16 <<<NQ * 8,      256, 0, stream>>>(qInV, qInG, WqInBf, DRVQ);
  wn_rows8_bf16<<<NQ*DRVQ/256, 256, 0, stream>>>(qOutV, qOutG, WqOutBf, NQ * DRVQ);
  cb_prep      <<<NQ*KCB/256,  256, 0, stream>>>(cb, cbT, cbn2);

  // zp = W_in @ z + b  -> act    (64 M-tiles = 2 blocks x 8 waves x 4 tiles)
  gemm_wn_bf16<<<dim3(2, NT, Bb), 256, 0, stream>>>(WinBf, z, inB, act, DIN);

  // Fused 32-step RVQ scan; act becomes qout in place
  rvq_fused<<<dim3(NT, Bb), 256, 0, stream>>>(ilen, WqInBf, qInB, WqOutBf, qOutB,
                                              cb, cbT, cbn2, act, idxo, cpart);

  // out = W_out @ qout + b
  gemm_wn_bf16<<<dim3(2, NT, Bb), 256, 0, stream>>>(WoutBf, act, outB, out, DRVQ);

  commit_reduce<<<(NQ * Bb + 255) / 256, 256, 0, stream>>>(cpart, como);
}
